// GCN2_8246337208546
// MI455X (gfx1250) — matrix operand order repaired and checked by reading, not compiled, not run on hardware
//
#include <hip/hip_runtime.h>
#include <math.h>

#define DH     128
#define DOUT   40
#define APITCH 136   // LDS pitch in halves: 272B stride -> conflict-free frag loads

typedef __bf16 bf16_t;
typedef bf16_t v16bf __attribute__((ext_vector_type(16)));
typedef float  v8f   __attribute__((ext_vector_type(8)));
typedef unsigned int v4u __attribute__((ext_vector_type(4)));

union FragU { v16bf v; v4u h[2]; };

__device__ __forceinline__ unsigned short f2bf(float f) {
  unsigned int u = __float_as_uint(f);
  u += 0x7FFFu + ((u >> 16) & 1u);     // round-to-nearest-even
  return (unsigned short)(u >> 16);
}

// ---------------------------------------------------------------- zero init
__global__ void k_zero(int* __restrict__ counts, float* __restrict__ stats, int n) {
  int i = blockIdx.x * 256 + threadIdx.x;
  if (i < n)   counts[i] = 0;
  if (i < 256) stats[i]  = 0.f;        // gsum[128] + gsq[128]
}

// ---------------------------------------------------------------- CSR build
__global__ void k_hist(const int* __restrict__ erow, int* __restrict__ counts, int e) {
  int i = blockIdx.x * 256 + threadIdx.x;
  if (i < e) atomicAdd(&counts[erow[i]], 1);
}

// single workgroup chained Hillis-Steele scan (runs once per launch)
__global__ __launch_bounds__(1024) void k_scan(const int* __restrict__ counts,
                                               int* __restrict__ rp,
                                               int* __restrict__ cursor, int n) {
  __shared__ int buf[2][1024];
  __shared__ int carry_s;
  int tid = threadIdx.x;
  if (tid == 0) carry_s = 0;
  __syncthreads();
  for (int base = 0; base < n; base += 1024) {
    int i = base + tid;
    int v = (i < n) ? counts[i] : 0;
    buf[0][tid] = v;
    __syncthreads();
    int src = 0;
    for (int off = 1; off < 1024; off <<= 1) {
      int t = buf[src][tid];
      if (tid >= off) t += buf[src][tid - off];
      buf[src ^ 1][tid] = t;
      __syncthreads();
      src ^= 1;
    }
    int incl  = buf[src][tid];
    int carry = carry_s;
    int excl  = carry + incl - v;
    if (i < n) { rp[i] = excl; cursor[i] = excl; }
    __syncthreads();
    if (tid == 1023) carry_s = carry + incl;
    __syncthreads();
  }
  if (tid == 0) rp[n] = carry_s;   // == E
}

__global__ void k_scatter(const int* __restrict__ erow, const int* __restrict__ ecol,
                          const float* __restrict__ ew, int* __restrict__ cursor,
                          int* __restrict__ col_s, float* __restrict__ w_s, int e) {
  int i = blockIdx.x * 256 + threadIdx.x;
  if (i >= e) return;
  int r   = erow[i];
  int pos = atomicAdd(&cursor[r], 1);
  col_s[pos] = ecol[i];
  w_s[pos]   = ew[i];
}

// ------------------------------------------------- SpMM fused with alpha mix
// one wave per row, lane = 4 consecutive features; a = 0.9*spmm(h) + 0.1*x0
__global__ __launch_bounds__(256) void k_spmm(const float* __restrict__ h,
                                              const float* __restrict__ x0,
                                              const int* __restrict__ rp,
                                              const int* __restrict__ col_s,
                                              const float* __restrict__ w_s,
                                              float* __restrict__ a, int nrows) {
  int lane = threadIdx.x & 31;
  int row  = blockIdx.x * 8 + (threadIdx.x >> 5);
  if (row >= nrows) return;
  int e0 = rp[row], e1 = rp[row + 1];
  int f  = lane * 4;
  float4 acc = make_float4(0.f, 0.f, 0.f, 0.f);
  for (int e = e0; e < e1; ++e) {
    int   c = col_s[e];            // wave-uniform -> scalar loads
    float w = w_s[e];
    float4 hv = *(const float4*)&h[(size_t)c * DH + f];
    acc.x += w * hv.x; acc.y += w * hv.y; acc.z += w * hv.z; acc.w += w * hv.w;
  }
  float4 xv = *(const float4*)&x0[(size_t)row * DH + f];
  float4 o;
  o.x = 0.9f * acc.x + 0.1f * xv.x;
  o.y = 0.9f * acc.y + 0.1f * xv.y;
  o.z = 0.9f * acc.z + 0.1f * xv.z;
  o.w = 0.9f * acc.w + 0.1f * xv.w;
  *(float4*)&a[(size_t)row * DH + f] = o;
}

// --------------------------------------------------------- WMMA frag helpers
__device__ __forceinline__ FragU load_fragA(const unsigned short* sA, int m, int kt, int lane) {
  // A 16x32 bf16: lane<16 holds K {kb..kb+7, kb+16..kb+23}, kb=kt*32; lane>=16 kb+=8
  FragU f;
  int kb = kt * 32 + ((lane >> 4) << 3);
  f.h[0] = *(const v4u*)&sA[m * APITCH + kb];
  f.h[1] = *(const v4u*)&sA[m * APITCH + kb + 16];
  return f;
}
__device__ __forceinline__ FragU load_fragB(const unsigned short* sB, int nt, int kt, int lane) {
  // B 32x16 bf16 (staged transposed [n][k]): lane col = lane&15, K {kb..kb+7, kb+16..kb+23}; lane>=16 kb+=8
  FragU f;
  int col = nt * 16 + (lane & 15);
  int ks  = kt * 32 + ((lane >> 4) << 3);
  f.h[0] = *(const v4u*)&sB[col * APITCH + ks];
  f.h[1] = *(const v4u*)&sB[col * APITCH + ks + 16];
  return f;
}

// ------------------------------------------------ input GEMM: x0 = relu(xW+b)
__global__ __launch_bounds__(256) void k_input_gemm(const float* __restrict__ Xg,
                                                    const float* __restrict__ Wg,
                                                    const float* __restrict__ bias,
                                                    float* __restrict__ x0,
                                                    float* __restrict__ hout, int nrows) {
  __shared__ alignas(16) unsigned short sA[DH * APITCH];
  __shared__ alignas(16) unsigned short sB[DH * APITCH];
  int tid = threadIdx.x, lane = tid & 31, wid = tid >> 5;
  int row0 = blockIdx.x * 128;
  for (int idx = tid; idx < DH * DH; idx += 256) {
    int r = idx >> 7, c = idx & 127;
    int row = row0 + r;
    float v = (row < nrows) ? Xg[(size_t)row * DH + c] : 0.f;
    sA[r * APITCH + c] = f2bf(v);
  }
  for (int idx = tid; idx < DH * DH; idx += 256) {
    int n = idx >> 7, k = idx & 127;
    sB[n * APITCH + k] = f2bf(Wg[k * DH + n]);    // stage transposed
  }
  __syncthreads();

  v8f acc[8];
  v8f z = {0.f,0.f,0.f,0.f,0.f,0.f,0.f,0.f};
  for (int i = 0; i < 8; ++i) acc[i] = z;
  int am = wid * 16 + (lane & 15);
  for (int kt = 0; kt < 4; ++kt) {
    FragU fa = load_fragA(sA, am, kt, lane);
    for (int nt = 0; nt < 8; ++nt) {
      FragU fb = load_fragB(sB, nt, kt, lane);
      acc[nt] = __builtin_amdgcn_wmma_f32_16x16x32_bf16(
          false, fa.v, false, fb.v, (short)0, acc[nt], false, false);
    }
  }
  for (int nt = 0; nt < 8; ++nt) {
    int n = nt * 16 + (lane & 15);
    float bv = bias[n];
    for (int r = 0; r < 8; ++r) {
      int m   = wid * 16 + r + ((lane >> 4) << 3);
      int row = row0 + m;
      if (row < nrows) {
        float v = acc[nt][r] + bv;
        v = v > 0.f ? v : 0.f;
        x0[(size_t)row * DH + n]   = v;
        hout[(size_t)row * DH + n] = v;
      }
    }
  }
}

// --------------- layer GEMM: a <- (1-beta)*a + beta*(a@W), fused BN statistics
__global__ __launch_bounds__(256) void k_layer_gemm(float* __restrict__ Ag,
                                                    const float* __restrict__ Wg,
                                                    float* __restrict__ gsum,
                                                    float* __restrict__ gsq,
                                                    float beta, int nrows) {
  __shared__ alignas(16) unsigned short sA[DH * APITCH];
  __shared__ alignas(16) unsigned short sB[DH * APITCH];
  __shared__ float sSum[DH];
  __shared__ float sSq[DH];
  int tid = threadIdx.x, lane = tid & 31, wid = tid >> 5;
  int row0 = blockIdx.x * 128;
  float beta1m = 1.f - beta;

  for (int idx = tid; idx < DH * DH; idx += 256) {
    int r = idx >> 7, c = idx & 127;
    int row = row0 + r;
    float v = (row < nrows) ? Ag[(size_t)row * DH + c] : 0.f;
    sA[r * APITCH + c] = f2bf(v);
  }
  for (int idx = tid; idx < DH * DH; idx += 256) {
    int n = idx >> 7, k = idx & 127;
    sB[n * APITCH + k] = f2bf(Wg[k * DH + n]);
  }
  if (tid < DH) { sSum[tid] = 0.f; sSq[tid] = 0.f; }
  __syncthreads();

  v8f acc[8];
  v8f z = {0.f,0.f,0.f,0.f,0.f,0.f,0.f,0.f};
  for (int i = 0; i < 8; ++i) acc[i] = z;
  int am = wid * 16 + (lane & 15);
  for (int kt = 0; kt < 4; ++kt) {
    FragU fa = load_fragA(sA, am, kt, lane);
    for (int nt = 0; nt < 8; ++nt) {
      FragU fb = load_fragB(sB, nt, kt, lane);
      acc[nt] = __builtin_amdgcn_wmma_f32_16x16x32_bf16(
          false, fa.v, false, fb.v, (short)0, acc[nt], false, false);
    }
  }
  // epilogue: mix with f32 a (re-read from L2), in-place store, BN stats
  for (int nt = 0; nt < 8; ++nt) {
    int n = nt * 16 + (lane & 15);
    float ssum = 0.f, ssq = 0.f;
    for (int r = 0; r < 8; ++r) {
      int m   = wid * 16 + r + ((lane >> 4) << 3);
      int row = row0 + m;
      if (row < nrows) {
        float av = Ag[(size_t)row * DH + n];
        float t  = beta1m * av + beta * acc[nt][r];
        Ag[(size_t)row * DH + n] = t;           // in-place: each elem owned by 1 lane
        ssum += t; ssq += t * t;
      }
    }
    atomicAdd(&sSum[n], ssum);                  // ds_add_f32
    atomicAdd(&sSq[n],  ssq);
  }
  __syncthreads();
  if (tid < DH) {
    atomicAdd(&gsum[tid], sSum[tid]);
    atomicAdd(&gsq[tid],  sSq[tid]);
  }
}

// ------------------------------------------------------------- BN finalize
__global__ void k_bn_finalize(float* __restrict__ gsum, float* __restrict__ gsq,
                              float* __restrict__ mu, float* __restrict__ rstd, float fN) {
  int c = threadIdx.x;
  float m   = gsum[c] / fN;
  float var = gsq[c] / fN - m * m;
  mu[c]   = m;
  rstd[c] = rsqrtf(var + 1e-5f);
  gsum[c] = 0.f;                 // re-arm accumulators for next layer
  gsq[c]  = 0.f;
}

// --------------------------------- h = relu(BN(t)*gamma + beta_bn + h)
__global__ __launch_bounds__(256) void k_bn_res_relu(const float* __restrict__ t,
                                                     float* __restrict__ h,
                                                     const float* __restrict__ mu,
                                                     const float* __restrict__ rstd,
                                                     const float* __restrict__ gam,
                                                     const float* __restrict__ bb, int nvec) {
  int i = blockIdx.x * 256 + threadIdx.x;
  if (i >= nvec) return;
  int c4 = (i & 31) * 4;
  float4 tv = ((const float4*)t)[i];
  float4 hv = ((const float4*)h)[i];
  float4 mv = *(const float4*)&mu[c4];
  float4 rv = *(const float4*)&rstd[c4];
  float4 gv = *(const float4*)&gam[c4];
  float4 bv = *(const float4*)&bb[c4];
  float4 o;
  o.x = (tv.x - mv.x) * rv.x * gv.x + bv.x + hv.x;
  o.y = (tv.y - mv.y) * rv.y * gv.y + bv.y + hv.y;
  o.z = (tv.z - mv.z) * rv.z * gv.z + bv.z + hv.z;
  o.w = (tv.w - mv.w) * rv.w * gv.w + bv.w + hv.w;
  o.x = o.x > 0.f ? o.x : 0.f;
  o.y = o.y > 0.f ? o.y : 0.f;
  o.z = o.z > 0.f ? o.z : 0.f;
  o.w = o.w > 0.f ? o.w : 0.f;
  ((float4*)h)[i] = o;
}

// -------------------------------------------- output GEMM: out = h@Wout + b
__global__ __launch_bounds__(256) void k_out_gemm(const float* __restrict__ Hg,
                                                  const float* __restrict__ Wg,
                                                  const float* __restrict__ bias,
                                                  float* __restrict__ out, int nrows) {
  __shared__ alignas(16) unsigned short sA[DH * APITCH];
  __shared__ alignas(16) unsigned short sB[48 * APITCH];
  int tid = threadIdx.x, lane = tid & 31, wid = tid >> 5;
  int row0 = blockIdx.x * 128;
  for (int idx = tid; idx < DH * DH; idx += 256) {
    int r = idx >> 7, c = idx & 127;
    int row = row0 + r;
    float v = (row < nrows) ? Hg[(size_t)row * DH + c] : 0.f;
    sA[r * APITCH + c] = f2bf(v);
  }
  for (int idx = tid; idx < 48 * DH; idx += 256) {
    int n = idx >> 7, k = idx & 127;             // n in [0,48)
    float v = (n < DOUT) ? Wg[k * DOUT + n] : 0.f;
    sB[n * APITCH + k] = f2bf(v);
  }
  __syncthreads();

  v8f acc[3];
  v8f z = {0.f,0.f,0.f,0.f,0.f,0.f,0.f,0.f};
  for (int i = 0; i < 3; ++i) acc[i] = z;
  int am = wid * 16 + (lane & 15);
  for (int kt = 0; kt < 4; ++kt) {
    FragU fa = load_fragA(sA, am, kt, lane);
    for (int nt = 0; nt < 3; ++nt) {
      FragU fb = load_fragB(sB, nt, kt, lane);
      acc[nt] = __builtin_amdgcn_wmma_f32_16x16x32_bf16(
          false, fa.v, false, fb.v, (short)0, acc[nt], false, false);
    }
  }
  for (int nt = 0; nt < 3; ++nt) {
    int n = nt * 16 + (lane & 15);
    if (n < DOUT) {
      float bv = bias[n];
      for (int r = 0; r < 8; ++r) {
        int m   = wid * 16 + r + ((lane >> 4) << 3);
        int row = row0 + m;
        if (row < nrows) out[(size_t)row * DOUT + n] = acc[nt][r] + bv;
      }
    }
  }
}

// =========================================================================
extern "C" void kernel_launch(void* const* d_in, const int* in_sizes, int n_in,
                              void* d_out, int out_size, void* d_ws, size_t ws_size,
                              hipStream_t stream) {
  const float* x    = (const float*)d_in[0];
  const float* ew   = (const float*)d_in[1];
  const float* Win  = (const float*)d_in[2];
  const float* bin  = (const float*)d_in[3];
  const float* Wc   = (const float*)d_in[4];
  const float* gam  = (const float*)d_in[5];
  const float* bet  = (const float*)d_in[6];
  const float* Wout = (const float*)d_in[7];
  const float* bout = (const float*)d_in[8];
  const int*   erow = (const int*)d_in[9];
  const int*   ecol = (const int*)d_in[10];
  float* out = (float*)d_out;

  const int N = in_sizes[0] / DH;
  const int E = in_sizes[9];

  char* ws = (char*)d_ws;
  size_t off = 0;
  auto wsa = [&](size_t bytes) -> char* {
    char* p = ws + off;
    off += (bytes + 255) & ~(size_t)255;
    return p;
  };
  float* x0     = (float*)wsa((size_t)N * DH * 4);
  float* h      = (float*)wsa((size_t)N * DH * 4);
  float* a      = (float*)wsa((size_t)N * DH * 4);
  int*   col_s  = (int*)  wsa((size_t)E * 4);
  float* w_s    = (float*)wsa((size_t)E * 4);
  int*   rp     = (int*)  wsa((size_t)(N + 1) * 4);
  int*   cursor = (int*)  wsa((size_t)N * 4);
  int*   counts = (int*)  wsa((size_t)N * 4);
  float* stats  = (float*)wsa(512 * 4);
  float* gsum = stats, *gsq = stats + 128, *mu = stats + 256, *rstd = stats + 384;

  // CSR build (once per launch; amortized over 16 layers)
  k_zero<<<(N + 255) / 256, 256, 0, stream>>>(counts, stats, N);
  k_hist<<<(E + 255) / 256, 256, 0, stream>>>(erow, counts, E);
  k_scan<<<1, 1024, 0, stream>>>(counts, rp, cursor, N);
  k_scatter<<<(E + 255) / 256, 256, 0, stream>>>(erow, ecol, ew, cursor, col_s, w_s, E);

  const int gb = (N + 127) / 128;
  k_input_gemm<<<gb, 256, 0, stream>>>(x, Win, bin, x0, h, N);

  for (int l = 1; l <= 16; ++l) {
    float beta = logf(0.5f / (float)l + 1.0f);
    k_spmm<<<(N + 7) / 8, 256, 0, stream>>>(h, x0, rp, col_s, w_s, a, N);
    k_layer_gemm<<<gb, 256, 0, stream>>>(a, Wc + (size_t)(l - 1) * DH * DH, gsum, gsq, beta, N);
    k_bn_finalize<<<1, 128, 0, stream>>>(gsum, gsq, mu, rstd, (float)N);
    k_bn_res_relu<<<(N * 32 + 255) / 256, 256, 0, stream>>>(
        a, h, mu, rstd, gam + (size_t)(l - 1) * DH, bet + (size_t)(l - 1) * DH, N * 32);
  }
  k_out_gemm<<<gb, 256, 0, stream>>>(h, Wout, bout, out, N);
}
